// Quantizer_1365799600669
// MI455X (gfx1250) — matrix operand order repaired and checked by reading, not compiled, not run on hardware
//
#include <hip/hip_runtime.h>

// ---------------------------------------------------------------------------
// VQ-VAE quantizer for MI455X (gfx1250, wave32, WMMA + TDM).
// d_in[0]: x [16,256,64,64] f32; d_in[1]: emb [1024,256] f32
// d_out:   [0] = c_loss, [1..16777216] = quant_out (NCHW f32)
// ---------------------------------------------------------------------------

typedef __attribute__((ext_vector_type(16))) __bf16       bf16x16;
typedef __attribute__((ext_vector_type(8)))  float        f32x8;
typedef __attribute__((ext_vector_type(4)))  unsigned int u32x4;
typedef __attribute__((ext_vector_type(8)))  unsigned int u32x8;
typedef __attribute__((ext_vector_type(8)))  int          i32x8;
typedef __attribute__((ext_vector_type(4)))  int          i32x4;

#define NUM_EMB   1024
#define EMB_DIM   256
#define HW        4096       // 64*64 pixels per image
#define NPIX      65536      // 16*4096 total rows
#define ROWS_PER_WAVE   16
#define WAVES_PER_BLOCK 4
#define MAIN_BLOCKS (NPIX / (ROWS_PER_WAVE * WAVES_PER_BLOCK)) // 1024

// TDM-staged B tile: 16 code rows x 256 bf16 = 512B/row, +16B LDS pad per row
// (pad_interval=6 -> every 128 DWORDs, pad_amount=3 -> 4 DWORDs) so per-lane
// ds_load_b128 fragment reads land on distinct bank groups (stride 132 DWORDs).
#define TDM_ROW_BYTES   528            // 512 data + 16 pad
#define TDM_BUF_BYTES   (16 * TDM_ROW_BYTES)

__device__ __forceinline__ unsigned short f2bf(float f) {
    unsigned u = __float_as_uint(f);
    unsigned r = u + 0x7FFFu + ((u >> 16) & 1u);   // round-to-nearest-even
    return (unsigned short)(r >> 16);
}

__device__ __forceinline__ unsigned lds_offset(const void* p) {
    // shared-aperture flat VA keeps the LDS byte offset in addr[31:0]
    return (unsigned)(unsigned long long)p;
}

// Issue one TDM DMA: 16x256 bf16 tile (row-major) -> LDS with row padding.
__device__ __forceinline__ void tdm_load_tile(const unsigned short* gsrc,
                                              unsigned lds_off) {
    unsigned long long ga = (unsigned long long)gsrc;
    u32x4 g0;
    g0[0] = 1u;                                   // count=1, user mode
    g0[1] = lds_off;                              // lds_addr
    g0[2] = (unsigned)ga;                         // global_addr[31:0]
    g0[3] = (unsigned)((ga >> 32) & 0x01FFFFFFu)  // global_addr[56:32]
          | (2u << 30);                           // type=2 ("image")
    i32x8 g1;
    g1[0] = (int)((1u << 16)       // data_size = 2 bytes
                | (1u << 20)       // pad_enable
                | (6u << 22)       // pad_interval: 128 DWORDs (one row)
                | (3u << 25));     // pad_amount:   4 DWORDs (16 B)
    g1[1] = (int)(256u << 16);     // tensor_dim0 = 256        (bits 79:48 lo)
    g1[2] = (int)(1024u << 16);    // dim0 hi=0 | tensor_dim1 = 1024 (lo)
    g1[3] = (int)(256u << 16);     // dim1 hi=0 | tile_dim0 = 256
    g1[4] = 16;                    // tile_dim1 = 16, tile_dim2 = 0
    g1[5] = 256;                   // tensor_dim0_stride = 256 elements
    g1[6] = 0;                     // stride hi | tensor_dim1_stride lo
    g1[7] = 0;
    i32x4 z4 = {0, 0, 0, 0};       // groups 2/3 unused (2-D tile)
    i32x8 z8 = {0, 0, 0, 0, 0, 0, 0, 0};
    __builtin_amdgcn_tensor_load_to_lds(g0, g1, z4, z4, z8, 0);
}

// --------------------------------------------------------------------------
// Kernel 1: codebook f32 -> bf16 + per-code squared norm
// --------------------------------------------------------------------------
__global__ __launch_bounds__(EMB_DIM)
void embprep(const float* __restrict__ emb,
             unsigned short* __restrict__ embbf,
             float* __restrict__ e2) {
    const int k = blockIdx.x;
    const int t = threadIdx.x;
    const float w = emb[(size_t)k * EMB_DIM + t];
    embbf[(size_t)k * EMB_DIM + t] = f2bf(w);

    float s = w * w;
    #pragma unroll
    for (int off = 16; off > 0; off >>= 1) s += __shfl_xor(s, off, 32);
    __shared__ float red[8];
    if ((t & 31) == 0) red[t >> 5] = s;
    __syncthreads();
    if (t == 0) {
        float tot = 0.f;
        #pragma unroll
        for (int i = 0; i < 8; ++i) tot += red[i];
        e2[k] = tot;
    }
}

// --------------------------------------------------------------------------
// Kernel 2: x [B,C,HW] f32 (channel-major) -> xT [NPIX][EMB_DIM] bf16
// 32x32 LDS tile transpose, coalesced both ways.
// --------------------------------------------------------------------------
__global__ __launch_bounds__(256)
void xpose(const float* __restrict__ x, unsigned short* __restrict__ xT) {
    __shared__ float tile[32][33];
    const int tx = threadIdx.x & 31;
    const int ty = threadIdx.x >> 5;     // 0..7
    const int p0 = blockIdx.x * 32;
    const int c0 = blockIdx.y * 32;
    const int b  = blockIdx.z;
    const float* xb = x + (size_t)b * EMB_DIM * HW;

    #pragma unroll
    for (int i = 0; i < 4; ++i) {
        const int c = ty + i * 8;
        tile[c][tx] = xb[(size_t)(c0 + c) * HW + p0 + tx];
    }
    __syncthreads();
    #pragma unroll
    for (int i = 0; i < 4; ++i) {
        const int pr = ty + i * 8;
        xT[(size_t)(b * HW + p0 + pr) * EMB_DIM + c0 + tx] = f2bf(tile[tx][pr]);
    }
}

// --------------------------------------------------------------------------
// Kernel 3: main — TDM-staged bf16 WMMA distance GEMM + argmin + gather + MSE
// One wave per 16-row tile; 4 waves (128 threads) per block share B via LDS.
// --------------------------------------------------------------------------
__global__ __launch_bounds__(WAVES_PER_BLOCK * 32)
void vq_main(const unsigned short* __restrict__ xT,
             const unsigned short* __restrict__ embbf,
             const float* __restrict__ e2,
             const float* __restrict__ x,
             const float* __restrict__ emb,
             float* __restrict__ qout,
             float* __restrict__ partials) {
    __shared__ __align__(64) unsigned char tdmbuf[2][TDM_BUF_BYTES];
    __shared__ int   idxTab[WAVES_PER_BLOCK][16];
    __shared__ float red[WAVES_PER_BLOCK];

    const int lane = threadIdx.x & 31;
    const int wave = threadIdx.x >> 5;
    const int m    = lane & 15;          // A row / B code column / D column
    const int hi   = lane >> 4;          // half-wave selector
    const int row_base = blockIdx.x * (ROWS_PER_WAVE * WAVES_PER_BLOCK)
                       + wave * ROWS_PER_WAVE;

    // Kick off DMA of code tile 0 while we fetch A fragments.
    if (wave == 0) tdm_load_tile(embbf, lds_offset(&tdmbuf[0][0]));

    // ---- A fragments: 16 rows x 256 K, resident in registers (8 x v16bf) ----
    // ISA 16-bit A 16x32 layout: lanes 0-15 hold K{0..7,16..23}, lanes 16-31
    // hold K{8..15,24..31}  => two aligned 16B loads per lane per chunk.
    bf16x16 afrag[8];
    const unsigned short* xrow = xT + (size_t)(row_base + m) * EMB_DIM;
    #pragma unroll
    for (int kc = 0; kc < 8; ++kc) {
        const int off = kc * 32 + hi * 8;
        const u32x4 a0 = *(const u32x4*)(xrow + off);
        const u32x4 a1 = *(const u32x4*)(xrow + off + 16);
        u32x8 u;
        #pragma unroll
        for (int j = 0; j < 4; ++j) { u[j] = a0[j]; u[j + 4] = a1[j]; }
        afrag[kc] = __builtin_bit_cast(bf16x16, u);
    }

    float best[8];
    int   bidx[8];
    #pragma unroll
    for (int v = 0; v < 8; ++v) { best[v] = 3.4e38f; bidx[v] = 0; }

    if (wave == 0) __builtin_amdgcn_s_wait_tensorcnt(0);
    __syncthreads();                     // tile 0 resident in tdmbuf[0]

    // ---- scan 64 code tiles of 16 codes each, double-buffered via TDM ----
    for (int ct = 0; ct < 64; ++ct) {
        if (wave == 0 && ct < 63)        // DMA next tile while computing
            tdm_load_tile(embbf + (size_t)(ct + 1) * 16 * EMB_DIM,
                          lds_offset(&tdmbuf[(ct + 1) & 1][0]));

        const int code = ct * 16 + m;    // this lane's code column
        const float e2v = e2[code];
        if (ct < 63)
            __builtin_prefetch(e2 + code + 16, 0, 1);

        // Preload ALL B fragments into distinct registers so the LDS pipe
        // runs ahead of the matrix pipe (no per-WMMA dscnt stall).
        // ISA 16-bit B 32x16 layout (matching A): lanes 0-15 = K{0..7,16..23},
        // lanes 16-31 = K{8..15,24..31} => two 16B regions per lane (two ds_load_b128,
        // bank-conflict-free thanks to the 16B row pad).
        const unsigned char* buf = &tdmbuf[ct & 1][0];
        bf16x16 bfrag[8];
        #pragma unroll
        for (int kc = 0; kc < 8; ++kc) {
            const unsigned char* brow = buf + m * TDM_ROW_BYTES
                                            + kc * 64 + hi * 16;
            const u32x4 b0 = *(const u32x4*)(brow);
            const u32x4 b1 = *(const u32x4*)(brow + 32);
            u32x8 u;
            #pragma unroll
            for (int j = 0; j < 4; ++j) { u[j] = b0[j]; u[j + 4] = b1[j]; }
            bfrag[kc] = __builtin_bit_cast(bf16x16, u);
        }

        // Two independent accumulator chains -> back-to-back WMMA issue
        // instead of serializing on the D->C RAW latency of one chain.
        f32x8 acc0 = {0.f, 0.f, 0.f, 0.f, 0.f, 0.f, 0.f, 0.f};
        f32x8 acc1 = {0.f, 0.f, 0.f, 0.f, 0.f, 0.f, 0.f, 0.f};
        #pragma unroll
        for (int kc = 0; kc < 8; kc += 2) {
            acc0 = __builtin_amdgcn_wmma_f32_16x16x32_bf16(
                     false, afrag[kc],     false, bfrag[kc],
                     (short)0, acc0, false, false);
            acc1 = __builtin_amdgcn_wmma_f32_16x16x32_bf16(
                     false, afrag[kc + 1], false, bfrag[kc + 1],
                     (short)0, acc1, false, false);
        }
        // D layout: lane L slot v -> (row = v + hi*8, code = ct*16 + (L&15))
        #pragma unroll
        for (int v = 0; v < 8; ++v) {
            const float s = e2v - 2.0f * (acc0[v] + acc1[v]);
            if (s < best[v]) { best[v] = s; bidx[v] = code; }
        }

        if (wave == 0) __builtin_amdgcn_s_wait_tensorcnt(0);
        __syncthreads();                 // publish ct+1, retire buffer ct
    }

    // ---- argmin reduce across the 16 code columns (xor butterfly, stays
    //      inside each half-wave; ties -> lower index like jnp.argmin) ----
    #pragma unroll
    for (int dm = 1; dm <= 8; dm <<= 1) {
        #pragma unroll
        for (int v = 0; v < 8; ++v) {
            const float ov = __shfl_xor(best[v], dm, 32);
            const int   oi = __shfl_xor(bidx[v], dm, 32);
            if (ov < best[v] || (ov == best[v] && oi < bidx[v])) {
                best[v] = ov; bidx[v] = oi;
            }
        }
    }

    if (m == 0) {
        #pragma unroll
        for (int v = 0; v < 8; ++v) idxTab[wave][hi * 8 + v] = bidx[v];
    }
    __syncthreads();

    // ---- gather codebook rows, write quant_out (NCHW), accumulate MSE ----
    const int b = row_base >> 12;            // /4096
    const int p = row_base & (HW - 1);
    const size_t base = (size_t)b * EMB_DIM * HW + p;   // + c*HW + pix
    const int myidx = idxTab[wave][m];
    const float* emrow = emb + (size_t)myidx * EMB_DIM;

    float errsq = 0.f;
    #pragma unroll 4
    for (int cc = 0; cc < 128; ++cc) {
        const int c = cc * 2 + hi;
        const float q = emrow[c];
        const size_t o = base + (size_t)c * HW + m;   // 16 lanes -> 64B segment
        const float xv = x[o];
        const float d = q - xv;
        errsq = fmaf(d, d, errsq);
        qout[o] = q;
    }

    #pragma unroll
    for (int off = 16; off > 0; off >>= 1) errsq += __shfl_xor(errsq, off, 32);
    if (lane == 0) red[wave] = errsq;
    __syncthreads();
    if (threadIdx.x == 0)
        partials[blockIdx.x] = red[0] + red[1] + red[2] + red[3];
}

// --------------------------------------------------------------------------
// Kernel 4: deterministic final reduction -> c_loss
// c_loss = (1 + COMM_COST*BETA) * sum_sq / (B*H*W*C)
// --------------------------------------------------------------------------
__global__ __launch_bounds__(256)
void vq_reduce(const float* __restrict__ partials, int n, float* __restrict__ out) {
    __shared__ float red[256];
    float s = 0.f;
    for (int i = threadIdx.x; i < n; i += 256) s += partials[i];
    red[threadIdx.x] = s;
    __syncthreads();
    #pragma unroll
    for (int k = 128; k > 0; k >>= 1) {
        if (threadIdx.x < k) red[threadIdx.x] += red[threadIdx.x + k];
        __syncthreads();
    }
    if (threadIdx.x == 0)
        out[0] = red[0] * (1.0625f / 16777216.0f);
}

// --------------------------------------------------------------------------
extern "C" void kernel_launch(void* const* d_in, const int* in_sizes, int n_in,
                              void* d_out, int out_size, void* d_ws, size_t ws_size,
                              hipStream_t stream) {
    const float* x   = (const float*)d_in[0];   // [16,256,64,64]
    const float* emb = (const float*)d_in[1];   // [1024,256]
    float* out = (float*)d_out;                 // [0]=loss, [1..]=quant NCHW

    char* ws = (char*)d_ws;
    unsigned short* xT    = (unsigned short*)ws;                        // 32 MB
    unsigned short* embbf = (unsigned short*)(ws + (size_t)NPIX * EMB_DIM * 2);
    float* e2       = (float*)((char*)embbf + (size_t)NUM_EMB * EMB_DIM * 2);
    float* partials = e2 + NUM_EMB;

    embprep<<<NUM_EMB, EMB_DIM, 0, stream>>>(emb, embbf, e2);
    xpose<<<dim3(HW / 32, EMB_DIM / 32, 16), 256, 0, stream>>>(x, xT);
    vq_main<<<MAIN_BLOCKS, WAVES_PER_BLOCK * 32, 0, stream>>>(
        xT, embbf, e2, x, emb, out + 1, partials);
    vq_reduce<<<1, 256, 0, stream>>>(partials, MAIN_BLOCKS, out);
}
